// Attention_46608985096563
// MI455X (gfx1250) — matrix operand order repaired and checked by reading, not compiled, not run on hardware
//
#include <hip/hip_runtime.h>

typedef __attribute__((ext_vector_type(16))) _Float16 v16h;
typedef __attribute__((ext_vector_type(8)))  _Float16 v8h;
typedef __attribute__((ext_vector_type(8)))  float    v8f;
typedef unsigned int u32x4 __attribute__((ext_vector_type(4)));
typedef int          i32x8 __attribute__((ext_vector_type(8)));
typedef int          i32x4 __attribute__((ext_vector_type(4)));

#define HEADS     4
#define DIM_HEAD  32
#define DIM       256
#define HIDDEN    128
#define NPIX      4096   // 64*64
#define BATCH     4
#define BH        16     // BATCH*HEADS
#define JBLK      64     // attention j-tile (columns per softmax round)

__device__ __forceinline__ int kbaseA(int j, int hf) {
    // 16-bit A-matrix 16x32, contiguous K order: vgpr j holds K pair hf*16 + 2j
    return hf * 16 + j * 2;
}

// WMMA D = A(16x32 f16) * B(32x16 f16) + C(16x16 f32)
__device__ __forceinline__ v8f wmma_f16(v16h a, v16h b, v8f c) {
    return __builtin_amdgcn_wmma_f32_16x16x32_f16(false, a, false, b,
                                                  (short)0, c, false, false);
}

// A-frag from a row-major [row][32] f16 segment: two contiguous 16B loads.
template <typename P>
__device__ __forceinline__ v16h load_a_row32(const P* row, int hf) {
    v16h a;
    v8h lo = *(const v8h*)(row + hf * 16);
    v8h hi = *(const v8h*)(row + hf * 16 + 8);
#pragma unroll
    for (int t = 0; t < 8; ++t) { a[t] = lo[t]; a[8 + t] = hi[t]; }
    return a;
}

// B-frag per lane = 16 contiguous halves (k = hf*16 + 2j, 2j+1).
template <typename P>
__device__ __forceinline__ v16h load_b_contig(const P* base) {
    v16h b;
    v8h lo = *(const v8h*)(base);
    v8h hi = *(const v8h*)(base + 8);
#pragma unroll
    for (int t = 0; t < 8; ++t) { b[t] = lo[t]; b[8 + t] = hi[t]; }
    return b;
}

// ---- Tensor Data Mover: 2D tile (elements are 2 bytes) global -> LDS ----
// D# packing per CDNA5 ISA ch.8: group0 = {flags, lds_addr, global_addr, type=2},
// group1 = {data_size=2B, tensor dims, tile dims, dim0 stride}.
__device__ __forceinline__ void tdm_load_2d(unsigned int lds_addr, const void* gptr,
                                            unsigned int tile_d0, unsigned int tile_d1,
                                            unsigned int tensor_d0, unsigned int tensor_d1,
                                            unsigned int stride0) {
    unsigned long long ga = (unsigned long long)gptr;
    u32x4 g0;
    g0[0] = 1u;                                         // count=1 (valid), user mode
    g0[1] = lds_addr;                                   // LDS byte address
    g0[2] = (unsigned int)ga;                           // global_addr[31:0]
    g0[3] = (unsigned int)((ga >> 32) & 0x01FFFFFFu) | 0x80000000u;  // addr[56:32] | type=2
    i32x8 g1;
    g1[0] = (int)(1u << 16);                            // data_size=1 -> 2 bytes
    g1[1] = (int)((tensor_d0 & 0xFFFFu) << 16);         // tensor_dim0[15:0] @ bits 63:48
    g1[2] = (int)((tensor_d0 >> 16) | ((tensor_d1 & 0xFFFFu) << 16));
    g1[3] = (int)((tensor_d1 >> 16) | (tile_d0 << 16)); // tile_dim0 @ bits 127:112
    g1[4] = (int)(tile_d1 & 0xFFFFu);                   // tile_dim1; tile_dim2=0
    g1[5] = (int)stride0;                               // tensor_dim0_stride[31:0]
    g1[6] = 0;
    g1[7] = 0;
    i32x4 z = {0, 0, 0, 0};
#if __clang_major__ >= 23
    i32x8 z8 = {0, 0, 0, 0, 0, 0, 0, 0};
    __builtin_amdgcn_tensor_load_to_lds(g0, g1, z, z, z8, 0);
#else
    __builtin_amdgcn_tensor_load_to_lds(g0, g1, z, z, 0);
#endif
}

// -------------------- Kernel 1: QKV projection --------------------
// qkv[o,p] = sum_c w_qkv[o,c] * x[b,c,p].  One 16x16 output tile per wave.
// Rows 0..127 -> Q (scaled by 32^-0.5 * log2(e)) [bh][pix][32];
// 128..255 -> K [bh][pix][32]; 256..383 -> V transposed [bh][d][pix].
__global__ __launch_bounds__(256) void qkv_kernel(
    const float* __restrict__ x, const float* __restrict__ wq,
    _Float16* __restrict__ qbuf, _Float16* __restrict__ kbuf,
    _Float16* __restrict__ vtbuf) {
    const int lane = threadIdx.x & 31;
    const int wave = threadIdx.x >> 5;
    const int n    = lane & 15;
    const int hf   = lane >> 4;

    const int p0 = (blockIdx.x * 8 + wave) * 16;   // pixel tile
    const int o0 = blockIdx.y * 16;                // out-channel tile
    const int b  = blockIdx.z;
    const float* xb = x + (size_t)b * DIM * NPIX;

    v8f acc = {};
#pragma unroll 1
    for (int c0 = 0; c0 < DIM; c0 += 32) {
        v16h af;
        const float* wrow = wq + (o0 + n) * DIM + c0;
#pragma unroll
        for (int j = 0; j < 8; ++j) {
            int k = kbaseA(j, hf);
            af[2 * j]     = (_Float16)wrow[k];
            af[2 * j + 1] = (_Float16)wrow[k + 1];
        }
        v16h bf;
        const float* xcol = xb + p0 + n;
#pragma unroll
        for (int j = 0; j < 8; ++j) {
            int c = c0 + hf * 16 + 2 * j;
            bf[2 * j]     = (_Float16)xcol[(size_t)c * NPIX];
            bf[2 * j + 1] = (_Float16)xcol[(size_t)(c + 1) * NPIX];
        }
        acc = wmma_f16(af, bf, acc);
    }

    const int section = o0 >> 7;          // 0=q 1=k 2=v
    const int within  = o0 & 127;
    const int h       = within >> 5;
    const int dbase   = within & 31;      // 0 or 16
    const int bh      = b * HEADS + h;
    const int p       = p0 + n;
    // 32^-0.5 * log2(e): softmax runs in base-2 (exp2) domain
    const float qscale = 0.17677669529663689f * 1.4426950408889634f;

    if (section == 0) {
        v8h o;
#pragma unroll
        for (int r = 0; r < 8; ++r) o[r] = (_Float16)(acc[r] * qscale);
        *(v8h*)(qbuf + ((size_t)bh * NPIX + p) * DIM_HEAD + dbase + hf * 8) = o;
    } else if (section == 1) {
        v8h o;
#pragma unroll
        for (int r = 0; r < 8; ++r) o[r] = (_Float16)acc[r];
        *(v8h*)(kbuf + ((size_t)bh * NPIX + p) * DIM_HEAD + dbase + hf * 8) = o;
    } else {
#pragma unroll
        for (int r = 0; r < 8; ++r) {
            int d = dbase + hf * 8 + r;
            vtbuf[((size_t)bh * DIM_HEAD + d) * NPIX + p] = (_Float16)acc[r];
        }
    }
}

// -------------------- Kernel 2: flash attention (TDM-staged) --------------------
// Block = 8 waves = 128 query rows of one (b,h). K/V j-tiles staged into LDS by
// the Tensor Data Mover, double-buffered; all 8 waves share them.
__global__ __launch_bounds__(256) void attn_kernel(
    const _Float16* __restrict__ qbuf, const _Float16* __restrict__ kbuf,
    const _Float16* __restrict__ vtbuf, _Float16* __restrict__ obuf) {
    __shared__ _Float16 Klds[2][JBLK][DIM_HEAD];   // 2 x 4KB  [j][d]
    __shared__ _Float16 Vlds[2][DIM_HEAD][JBLK];   // 2 x 4KB  [d][j]
    __shared__ _Float16 Plds[8][16][JBLK];         // 16KB, per-wave P tiles

    const int lane = threadIdx.x & 31;
    const int wave = threadIdx.x >> 5;
    const int n    = lane & 15;
    const int hf   = lane >> 4;

    const int bh = blockIdx.y;
    const int i0 = (blockIdx.x * 8 + wave) * 16;

    const _Float16* kbh = kbuf  + (size_t)bh * NPIX * DIM_HEAD;
    const _Float16* vbh = vtbuf + (size_t)bh * DIM_HEAD * NPIX;

    // Q A-frag, loaded once (K dim = dim_head = 32)
    const v16h qa = load_a_row32(qbuf + ((size_t)bh * NPIX + i0 + n) * DIM_HEAD, hf);

    v8f acc0 = {}, acc1 = {};                 // out cols d=0..15 / 16..31
    float mrow[8], lrow[8];
#pragma unroll
    for (int r = 0; r < 8; ++r) { mrow[r] = -1e30f; lrow[r] = 0.0f; }

    const int NT = NPIX / JBLK;               // 64 j-tiles

    // Prologue: stage tile 0
    if (wave == 0) {
        tdm_load_2d((unsigned int)(size_t)&Klds[0][0][0], kbh,
                    JBLK * DIM_HEAD, 1, 1u << 20, 1, JBLK * DIM_HEAD);
        tdm_load_2d((unsigned int)(size_t)&Vlds[0][0][0], vbh,
                    JBLK, DIM_HEAD, NPIX, DIM_HEAD, NPIX);
    }

#pragma unroll 1
    for (int jb = 0; jb < NT; ++jb) {
        const int cur = jb & 1;
        if (wave == 0) {
            if (jb + 1 < NT) {
                const int nxt = (jb + 1) & 1;
                const int j1  = (jb + 1) * JBLK;
                tdm_load_2d((unsigned int)(size_t)&Klds[nxt][0][0],
                            kbh + (size_t)j1 * DIM_HEAD,
                            JBLK * DIM_HEAD, 1, 1u << 20, 1, JBLK * DIM_HEAD);
                tdm_load_2d((unsigned int)(size_t)&Vlds[nxt][0][0],
                            vbh + j1, JBLK, DIM_HEAD, NPIX, DIM_HEAD, NPIX);
                __builtin_amdgcn_s_wait_tensorcnt((short)2);  // tile jb complete
            } else {
                __builtin_amdgcn_s_wait_tensorcnt((short)0);
            }
        }
        __syncthreads();   // publish Klds/Vlds[cur]

        // sim = Q * K^T over 64 columns: 4 WMMAs
        v8f z = {};
        v8f s0 = wmma_f16(qa, load_b_contig(&Klds[cur][ 0 + n][hf * 16]), z);
        v8f s1 = wmma_f16(qa, load_b_contig(&Klds[cur][16 + n][hf * 16]), z);
        v8f s2 = wmma_f16(qa, load_b_contig(&Klds[cur][32 + n][hf * 16]), z);
        v8f s3 = wmma_f16(qa, load_b_contig(&Klds[cur][48 + n][hf * 16]), z);

        // Online softmax (base 2) per row; rows m = hf*8+r across the 16-lane half
#pragma unroll
        for (int r = 0; r < 8; ++r) {
            float mx = fmaxf(fmaxf(s0[r], s1[r]), fmaxf(s2[r], s3[r]));
#pragma unroll
            for (int off = 1; off < 16; off <<= 1)
                mx = fmaxf(mx, __shfl_xor(mx, off, 32));
            float mnew = fmaxf(mrow[r], mx);
            float f    = exp2f(mrow[r] - mnew);
            float p0   = exp2f(s0[r] - mnew);
            float p1   = exp2f(s1[r] - mnew);
            float p2   = exp2f(s2[r] - mnew);
            float p3   = exp2f(s3[r] - mnew);
            float rs   = (p0 + p1) + (p2 + p3);
#pragma unroll
            for (int off = 1; off < 16; off <<= 1)
                rs += __shfl_xor(rs, off, 32);
            lrow[r] = lrow[r] * f + rs;
            mrow[r] = mnew;
            acc0[r] *= f;
            acc1[r] *= f;
            int m = hf * 8 + r;
            Plds[wave][m][n]      = (_Float16)p0;
            Plds[wave][m][16 + n] = (_Float16)p1;
            Plds[wave][m][32 + n] = (_Float16)p2;
            Plds[wave][m][48 + n] = (_Float16)p3;
        }
        __builtin_amdgcn_wave_barrier();

        // P A-frags (16x64 -> two 16x32) via LDS transpose-read
        v16h pa0 = load_a_row32(&Plds[wave][n][0], hf);
        v16h pa1 = load_a_row32(&Plds[wave][n][32], hf);
        __builtin_amdgcn_wave_barrier();

        // PV: 4 WMMAs
        acc0 = wmma_f16(pa0, load_b_contig(&Vlds[cur][n][hf * 16]), acc0);
        acc0 = wmma_f16(pa1, load_b_contig(&Vlds[cur][n][32 + hf * 16]), acc0);
        acc1 = wmma_f16(pa0, load_b_contig(&Vlds[cur][16 + n][hf * 16]), acc1);
        acc1 = wmma_f16(pa1, load_b_contig(&Vlds[cur][16 + n][32 + hf * 16]), acc1);

        __syncthreads();   // everyone done with Klds/Vlds[cur]
    }

    // Epilogue: normalize, store O as [b][h*32+d][pixel] f16
    const int b = bh / HEADS, h = bh % HEADS;
    v8h o0, o1;
#pragma unroll
    for (int r = 0; r < 8; ++r) {
        float inv = 1.0f / lrow[r];
        o0[r] = (_Float16)(acc0[r] * inv);
        o1[r] = (_Float16)(acc1[r] * inv);
    }
    size_t base0 = ((size_t)b * HIDDEN + h * DIM_HEAD + n)      * NPIX + i0 + hf * 8;
    size_t base1 = ((size_t)b * HIDDEN + h * DIM_HEAD + 16 + n) * NPIX + i0 + hf * 8;
    *(v8h*)(obuf + base0) = o0;
    *(v8h*)(obuf + base1) = o1;
}

// -------------------- Kernel 3: output projection --------------------
// y[b,oc,p] = sum_hd w_out[oc,hd] * O[b,hd,p] + b_out[oc]
__global__ __launch_bounds__(256) void proj_kernel(
    const _Float16* __restrict__ obuf, const float* __restrict__ wout,
    const float* __restrict__ bout, float* __restrict__ y) {
    const int lane = threadIdx.x & 31;
    const int wave = threadIdx.x >> 5;
    const int n    = lane & 15;
    const int hf   = lane >> 4;

    const int p0 = (blockIdx.x * 8 + wave) * 16;
    const int o0 = blockIdx.y * 16;
    const int b  = blockIdx.z;
    const _Float16* ob = obuf + (size_t)b * HIDDEN * NPIX;

    v8f acc = {};
#pragma unroll 1
    for (int k0 = 0; k0 < HIDDEN; k0 += 32) {
        v16h af;
        const float* wrow = wout + (o0 + n) * HIDDEN + k0;
#pragma unroll
        for (int j = 0; j < 8; ++j) {
            int k = kbaseA(j, hf);
            af[2 * j]     = (_Float16)wrow[k];
            af[2 * j + 1] = (_Float16)wrow[k + 1];
        }
        v16h bf;
        const _Float16* ocol = ob + p0 + n;
#pragma unroll
        for (int j = 0; j < 8; ++j) {
            int k = k0 + hf * 16 + 2 * j;
            bf[2 * j]     = ocol[(size_t)k * NPIX];
            bf[2 * j + 1] = ocol[(size_t)(k + 1) * NPIX];
        }
        acc = wmma_f16(af, bf, acc);
    }

    const int p = p0 + n;
#pragma unroll
    for (int r = 0; r < 8; ++r) {
        int oc = o0 + hf * 8 + r;
        y[((size_t)b * DIM + oc) * NPIX + p] = acc[r] + bout[oc];
    }
}

extern "C" void kernel_launch(void* const* d_in, const int* in_sizes, int n_in,
                              void* d_out, int out_size, void* d_ws, size_t ws_size,
                              hipStream_t stream) {
    (void)in_sizes; (void)n_in; (void)out_size; (void)ws_size;
    const float* x     = (const float*)d_in[0];
    const float* w_qkv = (const float*)d_in[1];
    const float* w_out = (const float*)d_in[2];
    const float* b_out = (const float*)d_in[3];
    float*       out   = (float*)d_out;

    char* ws = (char*)d_ws;
    const size_t SEG = (size_t)BH * NPIX * DIM_HEAD * sizeof(_Float16); // 4 MiB
    _Float16* qbuf = (_Float16*)(ws + 0 * SEG);
    _Float16* kbuf = (_Float16*)(ws + 1 * SEG);
    _Float16* vt   = (_Float16*)(ws + 2 * SEG);
    _Float16* obuf = (_Float16*)(ws + 3 * SEG);

    qkv_kernel<<<dim3(NPIX / 128, 24, BATCH), 256, 0, stream>>>(x, w_qkv, qbuf, kbuf, vt);
    attn_kernel<<<dim3(NPIX / 128, BH), 256, 0, stream>>>(qbuf, kbuf, vt, obuf);
    proj_kernel<<<dim3(NPIX / 128, DIM / 16, BATCH), 256, 0, stream>>>(obuf, w_out, b_out, out);
}
